// GAT_84988812853439
// MI455X (gfx1250) — compile-verified, matrix operand order checked by reading, not run on hardware
//
#include <hip/hip_runtime.h>
#include <hip/hip_bf16.h>
#include <stddef.h>

// ---------------- constants ----------------
#define HCDIM 256          // all conv layers produce heads*out = 256 features
#define GNUM 512
#define FC1_DIM 1024
#define NCLS_DIM 128
#define BN_RSQ 0.9999950000374997f   // 1/sqrt(1 + 1e-5)
#define LDS_STRIDE 112               // bytes per tile row in LDS (16B aligned, bank-friendly)

typedef __attribute__((ext_vector_type(16))) __bf16          v16bf;
typedef __attribute__((ext_vector_type(8)))  unsigned int    v8u;
typedef __attribute__((ext_vector_type(4)))  unsigned int    v4u;
typedef __attribute__((ext_vector_type(8)))  float           v8f;

// ---------------- device helpers ----------------
__device__ __forceinline__ unsigned short f2bf(float f) {
  unsigned u = __float_as_uint(f);
  u += 0x7FFFu + ((u >> 16) & 1u);       // round-to-nearest-even
  return (unsigned short)(u >> 16);
}
// order-preserving float -> uint key for atomic max
__device__ __forceinline__ unsigned fkey(float f) {
  unsigned u = __float_as_uint(f);
  return (u & 0x80000000u) ? ~u : (u | 0x80000000u);
}
__device__ __forceinline__ float fdecode(unsigned k) {
  unsigned u = (k & 0x80000000u) ? (k ^ 0x80000000u) : ~k;
  return __uint_as_float(u);
}

// ---------------- fills / converts ----------------
__global__ void fill_f32(float* p, float v, size_t n) {
  size_t i = (size_t)blockIdx.x * blockDim.x + threadIdx.x;
  if (i < n) p[i] = v;
}
__global__ void fill_u32(unsigned* p, unsigned v, size_t n) {
  size_t i = (size_t)blockIdx.x * blockDim.x + threadIdx.x;
  if (i < n) p[i] = v;
}
__global__ void cvt_bf16(const float* __restrict__ X, unsigned short* __restrict__ Y, size_t n) {
  size_t i = (size_t)blockIdx.x * blockDim.x + threadIdx.x;
  if (i < n) Y[i] = f2bf(X[i]);
}
// W[fin,fout] f32 -> Wt[fout,fin] bf16 (K-contiguous per output column)
__global__ void cvt_w_transpose(const float* __restrict__ W, unsigned short* __restrict__ Wt,
                                int fin, int fout) {
  size_t i = (size_t)blockIdx.x * blockDim.x + threadIdx.x;
  if (i >= (size_t)fin * fout) return;
  int k = (int)(i / fout), c = (int)(i % fout);
  Wt[(size_t)c * fin + k] = f2bf(W[i]);
}

// ---------------- WMMA GEMM: Y = act(Xbf[nrows,fin] @ W[fin,fout] + bias) ----------------
// Wt is the bf16-transposed weight [fout, fin]. 256 thr = 8 wave32; tile 32x64; K-step 32.
// nrows multiple of 32; fin multiple of 32; fout multiple of 64 at every call site.
template <int ACT>
__global__ __launch_bounds__(256)
void wmma_gemm_bf16(const unsigned short* __restrict__ X,
                    const unsigned short* __restrict__ Wt,
                    const float* __restrict__ bias, float* __restrict__ Y,
                    int nrows, int fin, int fout)
{
  __shared__ __attribute__((aligned(16))) unsigned short As[32][LDS_STRIDE / 2];
  __shared__ __attribute__((aligned(16))) unsigned short Bs[64][LDS_STRIDE / 2];

  const int tid  = threadIdx.x;
  const int row0 = blockIdx.x * 32;
  const int col0 = blockIdx.y * 64;
  const int wave = tid >> 5, lane = tid & 31;
  const int wm   = wave >> 2, wn = wave & 3;
  const int rsel = lane & 15, hsel = lane >> 4;

  // async staging: each thread moves 8B of A (32x64B = 2KB) and 16B of B (64x64B = 4KB)
  const int aoff = tid * 8;   const int ar = aoff >> 6, ak = aoff & 63;
  const int boff = tid * 16;  const int bc = boff >> 6, bk = boff & 63;
  const unsigned lA = (unsigned)(size_t)((char*)&As[0][0] + ar * LDS_STRIDE + ak);
  const unsigned lB = (unsigned)(size_t)((char*)&Bs[0][0] + bc * LDS_STRIDE + bk);

  v8f acc = {};
  for (int k0 = 0; k0 < fin; k0 += 32) {
    unsigned long long gA = (unsigned long long)(size_t)
        ((const char*)X + ((size_t)(row0 + ar) * fin + k0) * 2 + ak);
    unsigned long long gB = (unsigned long long)(size_t)
        ((const char*)Wt + ((size_t)(col0 + bc) * fin + k0) * 2 + bk);
    asm volatile("global_load_async_to_lds_b64 %0, %1, off" :: "v"(lA), "v"(gA) : "memory");
    asm volatile("global_load_async_to_lds_b128 %0, %1, off" :: "v"(lB), "v"(gB) : "memory");
    asm volatile("s_wait_asynccnt 0x0" ::: "memory");
    __syncthreads();

    // ISA 7.12.2 bf16 fragments: two contiguous 16B chunks per matrix per lane
    const v4u* pa = (const v4u*)((const char*)&As[0][0] + (wm * 16 + rsel) * LDS_STRIDE + hsel * 16);
    const v4u* pb = (const v4u*)((const char*)&Bs[0][0] + (wn * 16 + rsel) * LDS_STRIDE + hsel * 16);
    v4u alo = pa[0], ahi = pa[2];   // +32B
    v4u blo = pb[0], bhi = pb[2];
    v8u au, bu;
#pragma unroll
    for (int i = 0; i < 4; ++i) {
      au[i] = alo[i]; au[4 + i] = ahi[i];
      bu[i] = blo[i]; bu[4 + i] = bhi[i];
    }
    acc = __builtin_amdgcn_wmma_f32_16x16x32_bf16(
        false, __builtin_bit_cast(v16bf, au),
        false, __builtin_bit_cast(v16bf, bu),
        (short)0, acc, false, false);
    __syncthreads();
  }

  const int col = col0 + wn * 16 + rsel;
  const float bv = bias[col];
  const int rbase = row0 + wm * 16 + hsel * 8;
#pragma unroll
  for (int v = 0; v < 8; ++v) {
    float o = acc[v] + bv;
    if (ACT) o = fmaxf(o, 0.f);
    Y[(size_t)(rbase + v) * fout + col] = o;
  }
}

// ---------------- edge kernels (GATv2 attention) ----------------
// edge e in [0,E): (src[e], dst[e]); e in [E,E+N): self loop (e-E, e-E)
__global__ void edge_score(const float* __restrict__ xl, const float* __restrict__ xr,
                           const float* __restrict__ att,
                           const int* __restrict__ src, const int* __restrict__ dst,
                           int E, int N, int H, int C,
                           float* __restrict__ score, unsigned* __restrict__ maxkey)
{
  size_t t = (size_t)blockIdx.x * blockDim.x + threadIdx.x;
  size_t ET = (size_t)(E + N) * H;
  if (t >= ET) return;
  int e = (int)(t / H), h = (int)(t % H);
  int s, d;
  if (e < E) { s = src[e]; d = dst[e]; } else { s = d = e - E; }
  const float4* pl = (const float4*)(xl + (size_t)s * HCDIM + h * C);
  const float4* pr = (const float4*)(xr + (size_t)d * HCDIM + h * C);
  const float4* pa = (const float4*)(att + h * C);
  float acc = 0.f;
  for (int c = 0; c < (C >> 2); ++c) {
    float4 a = pl[c], b = pr[c], w = pa[c];
    float v0 = a.x + b.x; v0 = (v0 > 0.f) ? v0 : 0.2f * v0; acc += v0 * w.x;
    float v1 = a.y + b.y; v1 = (v1 > 0.f) ? v1 : 0.2f * v1; acc += v1 * w.y;
    float v2 = a.z + b.z; v2 = (v2 > 0.f) ? v2 : 0.2f * v2; acc += v2 * w.z;
    float v3 = a.w + b.w; v3 = (v3 > 0.f) ? v3 : 0.2f * v3; acc += v3 * w.w;
  }
  score[t] = acc;
  atomicMax(&maxkey[(size_t)d * H + h], fkey(acc));
}

__global__ void edge_softmax(float* __restrict__ score, const unsigned* __restrict__ maxkey,
                             float* __restrict__ den,
                             const int* __restrict__ dst, int E, int N, int H)
{
  size_t t = (size_t)blockIdx.x * blockDim.x + threadIdx.x;
  size_t ET = (size_t)(E + N) * H;
  if (t >= ET) return;
  int e = (int)(t / H), h = (int)(t % H);
  int d = (e < E) ? dst[e] : (e - E);
  float m = fdecode(maxkey[(size_t)d * H + h]);
  float a = __expf(score[t] - m);
  score[t] = a;
  atomicAdd(&den[(size_t)d * H + h], a);
}

__global__ void edge_aggregate(const float* __restrict__ xl, const float* __restrict__ score,
                               const float* __restrict__ den,
                               const int* __restrict__ src, const int* __restrict__ dst,
                               int E, int N, int H, int C, float* __restrict__ agg)
{
  size_t t = (size_t)blockIdx.x * blockDim.x + threadIdx.x;
  size_t ET = (size_t)(E + N) * H;
  if (t >= ET) return;
  int e = (int)(t / H), h = (int)(t % H);
  int s, d;
  if (e < E) { s = src[e]; d = dst[e]; } else { s = d = e - E; }
  float alpha = score[t] / den[(size_t)d * H + h];
  const float4* pl = (const float4*)(xl + (size_t)s * HCDIM + h * C);
  float* po = agg + (size_t)d * HCDIM + h * C;
  for (int c = 0; c < (C >> 2); ++c) {
    float4 a = pl[c];
    atomicAdd(&po[4 * c + 0], a.x * alpha);
    atomicAdd(&po[4 * c + 1], a.y * alpha);
    atomicAdd(&po[4 * c + 2], a.z * alpha);
    atomicAdd(&po[4 * c + 3], a.w * alpha);
  }
}

// out[n,c] = elu(bn(mean_h agg + bias)) + lin[n,c]   (H=1 => concat path)
__global__ void post_kernel(const float* __restrict__ agg, const float* __restrict__ bias,
                            const float* __restrict__ g, const float* __restrict__ b,
                            const float* __restrict__ lin, float* __restrict__ out,
                            int N, int H, int C)
{
  size_t i = (size_t)blockIdx.x * blockDim.x + threadIdx.x;
  if (i >= (size_t)N * C) return;
  int n = (int)(i / C), c = (int)(i % C);
  float v = 0.f;
  for (int h = 0; h < H; ++h) v += agg[(size_t)n * HCDIM + h * C + c];
  v = v / (float)H + bias[c];
  v = g[c] * (v * BN_RSQ) + b[c];
  v = (v > 0.f) ? v : (__expf(v) - 1.f);   // ELU
  out[i] = v + lin[i];
}

// ---------------- pooling ----------------
__global__ void pool_accum(const float* __restrict__ h, const int* __restrict__ batch,
                           float* __restrict__ sums, int N)
{
  size_t i = (size_t)blockIdx.x * blockDim.x + threadIdx.x;
  if (i >= (size_t)N * HCDIM) return;
  int n = (int)(i / HCDIM), c = (int)(i % HCDIM);
  atomicAdd(&sums[(size_t)batch[n] * HCDIM + c], h[i]);
}
__global__ void pool_count(const int* __restrict__ batch, float* __restrict__ cnt, int N)
{
  int n = blockIdx.x * blockDim.x + threadIdx.x;
  if (n < N) atomicAdd(&cnt[batch[n]], 1.f);
}
__global__ void pool_final(const float* __restrict__ sums, const float* __restrict__ cnt,
                           float* __restrict__ pooled)
{
  size_t i = (size_t)blockIdx.x * blockDim.x + threadIdx.x;
  if (i >= (size_t)GNUM * HCDIM) return;
  pooled[i] = sums[i] / fmaxf(cnt[i / HCDIM], 1.f);
}

// ---------------- input index map (JAX pytree = sorted dict keys) ----------------
enum {
  IN_X = 0,
  IN_BN1_B, IN_BN1_G, IN_BN2_B, IN_BN2_G, IN_BN3_B, IN_BN3_G,
  IN_BN4_B, IN_BN4_G, IN_BN5_B, IN_BN5_G,                        // 1..10
  IN_C1_WL, IN_C1_WR, IN_C1_ATT, IN_C1_BIAS, IN_C1_BL, IN_C1_BR, // 11..16
  IN_C2_WL, IN_C2_WR, IN_C2_ATT, IN_C2_BIAS, IN_C2_BL, IN_C2_BR, // 17..22
  IN_C3_WL, IN_C3_WR, IN_C3_ATT, IN_C3_BIAS, IN_C3_BL, IN_C3_BR, // 23..28
  IN_C5_WL, IN_C5_WR, IN_C5_ATT, IN_C5_BIAS, IN_C5_BL, IN_C5_BR, // 29..34
  IN_FC1_W, IN_FC1_B, IN_FC2_W, IN_FC2_B,                        // 35..38
  IN_LIN1_W, IN_LIN1_B, IN_LIN2_W, IN_LIN2_B, IN_LIN3_W, IN_LIN3_B,
  IN_LIN4_W, IN_LIN4_B, IN_LIN5_W, IN_LIN5_B,                    // 39..48
  IN_EI, IN_BATCH                                                // 49,50
};

struct ConvP { const float *Wl, *bl, *Wr, *br, *att, *bias; };

static void gemm_bf16(const unsigned short* XB, const float* W, unsigned short* WT,
                      const float* bias, float* Y, int nrows, int fin, int fout,
                      int act, hipStream_t s)
{
  size_t wsz = (size_t)fin * fout;
  cvt_w_transpose<<<(unsigned)((wsz + 255) / 256), 256, 0, s>>>(W, WT, fin, fout);
  dim3 g(nrows / 32, fout / 64);
  if (act)
    wmma_gemm_bf16<1><<<g, 256, 0, s>>>(XB, WT, bias, Y, nrows, fin, fout);
  else
    wmma_gemm_bf16<0><<<g, 256, 0, s>>>(XB, WT, bias, Y, nrows, fin, fout);
}

static void run_gat_layer(const float* xin, int fin, ConvP cp, int H, int C,
                          const float* bng, const float* bnb,
                          const float* linW, const float* linb, int linF,
                          int postH, int postC,
                          unsigned short* XB, unsigned short* WT,
                          float* XL, float* XR, float* AGG, float* LINB,
                          float* SC, unsigned* MK, float* DEN, float* OUT,
                          const int* src, const int* dst, int N, int E, hipStream_t s)
{
  size_t nx = (size_t)N * fin;
  cvt_bf16<<<(unsigned)((nx + 255) / 256), 256, 0, s>>>(xin, XB, nx);
  gemm_bf16(XB, cp.Wl, WT, cp.bl, XL, N, fin, HCDIM, 0, s);
  gemm_bf16(XB, cp.Wr, WT, cp.br, XR, N, fin, HCDIM, 0, s);
  gemm_bf16(XB, linW, WT, linb, LINB, N, fin, linF, 0, s);

  size_t nh = (size_t)N * H;
  fill_u32<<<(unsigned)((nh + 255) / 256), 256, 0, s>>>(MK, 0u, nh);
  fill_f32<<<(unsigned)((nh + 255) / 256), 256, 0, s>>>(DEN, 0.f, nh);
  size_t na = (size_t)N * HCDIM;
  fill_f32<<<(unsigned)((na + 255) / 256), 256, 0, s>>>(AGG, 0.f, na);

  size_t eh = (size_t)(E + N) * H;
  unsigned eb = (unsigned)((eh + 255) / 256);
  edge_score<<<eb, 256, 0, s>>>(XL, XR, cp.att, src, dst, E, N, H, C, SC, MK);
  edge_softmax<<<eb, 256, 0, s>>>(SC, MK, DEN, dst, E, N, H);
  edge_aggregate<<<eb, 256, 0, s>>>(XL, SC, DEN, src, dst, E, N, H, C, AGG);

  size_t np = (size_t)N * postC;
  post_kernel<<<(unsigned)((np + 255) / 256), 256, 0, s>>>(AGG, cp.bias, bng, bnb, LINB, OUT,
                                                           N, postH, postC);
}

extern "C" void kernel_launch(void* const* d_in, const int* in_sizes, int n_in,
                              void* d_out, int out_size, void* d_ws, size_t ws_size,
                              hipStream_t stream)
{
  if (n_in < 51) return;
  const float* x = (const float*)d_in[IN_X];
  const int N = in_sizes[IN_X] / 64;
  const int E = in_sizes[IN_EI] / 2;
  const int* src   = (const int*)d_in[IN_EI];
  const int* dst   = src + E;
  const int* batch = (const int*)d_in[IN_BATCH];

#define P(i) ((const float*)d_in[i])
  ConvP c1 = {P(IN_C1_WL), P(IN_C1_BL), P(IN_C1_WR), P(IN_C1_BR), P(IN_C1_ATT), P(IN_C1_BIAS)};
  ConvP c2 = {P(IN_C2_WL), P(IN_C2_BL), P(IN_C2_WR), P(IN_C2_BR), P(IN_C2_ATT), P(IN_C2_BIAS)};
  ConvP c3 = {P(IN_C3_WL), P(IN_C3_BL), P(IN_C3_WR), P(IN_C3_BR), P(IN_C3_ATT), P(IN_C3_BIAS)};
  ConvP c5 = {P(IN_C5_WL), P(IN_C5_BL), P(IN_C5_WR), P(IN_C5_BR), P(IN_C5_ATT), P(IN_C5_BIAS)};

  // workspace layout (in floats)
  float* ws = (float*)d_ws;
  size_t off = 0;
  auto alloc = [&](size_t nfloats) { float* p = ws + off; off += nfloats; return p; };
  float*    XL   = alloc((size_t)N * HCDIM);
  float*    XR   = alloc((size_t)N * HCDIM);
  float*    AGG  = alloc((size_t)N * HCDIM);
  float*    LINB = alloc((size_t)N * HCDIM);
  float*    SC   = alloc((size_t)(E + N) * 4);
  unsigned* MK   = (unsigned*)alloc((size_t)N * 4);
  float*    DEN  = alloc((size_t)N * 4);
  float*    H1   = alloc((size_t)N * HCDIM);
  float*    H2   = alloc((size_t)N * 64);
  float*    H3   = alloc((size_t)N * 64);
  float*    H4   = alloc((size_t)N * 64);
  float*    H5   = alloc((size_t)N * HCDIM);
  float*    SUM  = alloc((size_t)GNUM * HCDIM);
  float*    CNT  = alloc((size_t)GNUM);
  float*    POOL = alloc((size_t)GNUM * HCDIM);
  float*    FC1O = alloc((size_t)GNUM * FC1_DIM);
  unsigned short* XB = (unsigned short*)alloc((size_t)N * HCDIM / 2);   // bf16 activations
  unsigned short* WT = (unsigned short*)alloc(262144 / 2);              // bf16 W^T (max 256x1024)
  if (off * sizeof(float) > ws_size) return;

  // layer 1: concat heads (post as H=1,C=256), fin 64
  run_gat_layer(x, 64, c1, 4, 64, P(IN_BN1_G), P(IN_BN1_B),
                P(IN_LIN1_W), P(IN_LIN1_B), 256, 1, 256,
                XB, WT, XL, XR, AGG, LINB, SC, MK, DEN, H1, src, dst, N, E, stream);
  // layer 2: mean heads, fin 256 -> 64
  run_gat_layer(H1, 256, c2, 4, 64, P(IN_BN2_G), P(IN_BN2_B),
                P(IN_LIN2_W), P(IN_LIN2_B), 64, 4, 64,
                XB, WT, XL, XR, AGG, LINB, SC, MK, DEN, H2, src, dst, N, E, stream);
  // layer 3
  run_gat_layer(H2, 64, c3, 4, 64, P(IN_BN3_G), P(IN_BN3_B),
                P(IN_LIN3_W), P(IN_LIN3_B), 64, 4, 64,
                XB, WT, XL, XR, AGG, LINB, SC, MK, DEN, H3, src, dst, N, E, stream);
  // layer 4 (reuses conv3 weights)
  run_gat_layer(H3, 64, c3, 4, 64, P(IN_BN4_G), P(IN_BN4_B),
                P(IN_LIN4_W), P(IN_LIN4_B), 64, 4, 64,
                XB, WT, XL, XR, AGG, LINB, SC, MK, DEN, H4, src, dst, N, E, stream);
  // layer 5: 1 head x 256
  run_gat_layer(H4, 64, c5, 1, 256, P(IN_BN5_G), P(IN_BN5_B),
                P(IN_LIN5_W), P(IN_LIN5_B), 256, 1, 256,
                XB, WT, XL, XR, AGG, LINB, SC, MK, DEN, H5, src, dst, N, E, stream);

  // global mean pool
  size_t ns = (size_t)GNUM * HCDIM;
  fill_f32<<<(unsigned)((ns + 255) / 256), 256, 0, stream>>>(SUM, 0.f, ns);
  fill_f32<<<(GNUM + 255) / 256, 256, 0, stream>>>(CNT, 0.f, (size_t)GNUM);
  size_t np = (size_t)N * HCDIM;
  pool_accum<<<(unsigned)((np + 255) / 256), 256, 0, stream>>>(H5, batch, SUM, N);
  pool_count<<<(N + 255) / 256, 256, 0, stream>>>(batch, CNT, N);
  pool_final<<<(unsigned)((ns + 255) / 256), 256, 0, stream>>>(SUM, CNT, POOL);

  // FC head
  size_t npool = (size_t)GNUM * HCDIM;
  cvt_bf16<<<(unsigned)((npool + 255) / 256), 256, 0, stream>>>(POOL, XB, npool);
  gemm_bf16(XB, P(IN_FC1_W), WT, P(IN_FC1_B), FC1O, GNUM, HCDIM, FC1_DIM, 1, stream);
  size_t nf1 = (size_t)GNUM * FC1_DIM;
  cvt_bf16<<<(unsigned)((nf1 + 255) / 256), 256, 0, stream>>>(FC1O, XB, nf1);
  gemm_bf16(XB, P(IN_FC2_W), WT, P(IN_FC2_B), (float*)d_out, GNUM, FC1_DIM, NCLS_DIM, 0, stream);
#undef P
}
